// LocalCausalSelfAttention_43284680409119
// MI455X (gfx1250) — matrix operand order repaired and checked by reading, not compiled, not run on hardware
//
#include <hip/hip_runtime.h>

// ---------------------------------------------------------------------------
// MI455X (gfx1250) attention block: bf16 WMMA GEMMs + flash attention.
// Wave32, WMMA 16x16x32 bf16 -> f32. Async global->LDS copies when available.
// ---------------------------------------------------------------------------

typedef __attribute__((ext_vector_type(16))) __bf16 v16bf;
typedef __attribute__((ext_vector_type(8)))  __bf16 v8bf;
typedef __attribute__((ext_vector_type(8)))  float  v8f;
typedef __attribute__((ext_vector_type(4)))  int    v4i;

#define M_TOT 8192      // B*NG*LS = 2*8*512
#define E_DIM 1024
#define E3    3072
#define LSEQ  512
#define NHEAD 16
#define HDIM  64

// gfx1250 async global->LDS copy (ASYNCcnt-tracked), guarded so the file
// compiles on toolchains that lack the builtins.
#if defined(__has_builtin)
#if __has_builtin(__builtin_amdgcn_global_load_async_to_lds_b128) && \
    __has_builtin(__builtin_amdgcn_s_wait_asynccnt)
#define USE_ASYNC_LDS 1
#endif
#endif

#if defined(USE_ASYNC_LDS)
typedef __attribute__((address_space(1))) v4i gas_v4i;  // global int4
typedef __attribute__((address_space(3))) v4i las_v4i;  // LDS int4
#endif

__device__ __forceinline__ void copy16_to_lds(__bf16* lptr, const __bf16* gptr) {
#if defined(USE_ASYNC_LDS)
  __builtin_amdgcn_global_load_async_to_lds_b128(
      (gas_v4i*)gptr, (las_v4i*)lptr, 0, 0);
#else
  *(uint4*)lptr = *(const uint4*)gptr;
#endif
}
__device__ __forceinline__ void wait_lds_copies() {
#if defined(USE_ASYNC_LDS)
  __builtin_amdgcn_s_wait_asynccnt(0);
#endif
}

__device__ __forceinline__ __bf16 f2bf(float f) {
  unsigned u = __builtin_bit_cast(unsigned, f);
  unsigned r = u + 0x7FFFu + ((u >> 16) & 1u);
  unsigned short h = (unsigned short)(r >> 16);
  return __builtin_bit_cast(__bf16, h);
}
__device__ __forceinline__ float bf2f(__bf16 b) {
  unsigned short h = __builtin_bit_cast(unsigned short, b);
  unsigned u = ((unsigned)h) << 16;
  return __builtin_bit_cast(float, u);
}

// A-fragment: 16x32 bf16 tile. lane L (L<16): row M=L, K in {0..7, 16..23};
// lane L>=16: row M=L-16, K in {8..15, 24..31}. (ISA 7.12.2)
__device__ __forceinline__ v16bf load_frag_a(const __bf16* p, int r0, int k0,
                                             int ld, int lane) {
  int row = lane & 15;
  int kb  = (lane >> 4) << 3;
  const __bf16* q = p + (size_t)(r0 + row) * ld + k0 + kb;
  union { v16bf v; v8bf h[2]; } u;
  u.h[0] = *(const v8bf*)(q);
  u.h[1] = *(const v8bf*)(q + 16);
  return u.v;
}

// B-fragment: 32x16 bf16 tile, sourced from B^T stored [N][K].
// lane L: col N=L%16, K in {8h..8h+7, 16+8h..16+8h+7}, h=L/16.
__device__ __forceinline__ v16bf load_frag_b(const __bf16* p, int n0, int k0,
                                             int ld, int lane) {
  int n  = lane & 15;
  int kb = (lane >> 4) << 3;
  const __bf16* q = p + (size_t)(n0 + n) * ld + k0 + kb;
  union { v16bf v; v8bf h[2]; } u;
  u.h[0] = *(const v8bf*)(q);
  u.h[1] = *(const v8bf*)(q + 16);
  return u.v;
}

__device__ __forceinline__ v8f wmma_bf16(v16bf a, v16bf b, v8f c) {
  return __builtin_amdgcn_wmma_f32_16x16x32_bf16(false, a, false, b, (short)0,
                                                 c, false, false);
}

__device__ __forceinline__ v8f v8f_zero() {
  v8f z = {0.f, 0.f, 0.f, 0.f, 0.f, 0.f, 0.f, 0.f};
  return z;
}

// ---------------------------------------------------------------------------
// 1) fp32 -> bf16 elementwise convert, 8 elems/thread
// ---------------------------------------------------------------------------
__global__ __launch_bounds__(256)
void cvt_f32_bf16_kernel(const float* __restrict__ src, __bf16* __restrict__ dst) {
  int i = (blockIdx.x * 256 + threadIdx.x) * 8;
  float4 a = *(const float4*)(src + i);
  float4 b = *(const float4*)(src + i + 4);
  union { uint4 u; __bf16 e[8]; } o;
  o.e[0] = f2bf(a.x); o.e[1] = f2bf(a.y); o.e[2] = f2bf(a.z); o.e[3] = f2bf(a.w);
  o.e[4] = f2bf(b.x); o.e[5] = f2bf(b.y); o.e[6] = f2bf(b.z); o.e[7] = f2bf(b.w);
  *(uint4*)(dst + i) = o.u;
}

// ---------------------------------------------------------------------------
// 2) W [K,N] fp32 -> Wt [N,K] bf16 (so WMMA B-fragments are contiguous)
// ---------------------------------------------------------------------------
__global__ __launch_bounds__(256)
void transpose_cvt_kernel(const float* __restrict__ W, __bf16* __restrict__ Wt,
                          int K, int N) {
  int i = blockIdx.x * 256 + threadIdx.x;
  if (i < K * N) {
    int k = i / N, n = i % N;
    Wt[(size_t)n * K + k] = f2bf(W[i]);
  }
}

// ---------------------------------------------------------------------------
// 3) WMMA GEMM: C[M,N] = A[M,K] * Bt[N,K]^T. Compile-time N,K and output
//    type (no runtime strides -> no 64-bit mults, no epilogue branches).
//    128x128 block tile, 8 waves, wave strip 32x64, K step 64.
// ---------------------------------------------------------------------------
template <int N, int K, bool F32OUT>
__global__ __launch_bounds__(256)
void wmma_gemm_kernel(const __bf16* __restrict__ A, const __bf16* __restrict__ Bt,
                      void* __restrict__ Cout) {
  __shared__ __bf16 sA[128 * 64];   // 16 KB
  __shared__ __bf16 sB[128 * 64];   // 16 KB

  const int tid  = threadIdx.x;
  const int lane = tid & 31;
  const int wave = tid >> 5;
  const int wm   = wave & 3;   // 32-row strip
  const int wn   = wave >> 2;  // 64-col strip
  const int bm   = blockIdx.y * 128;
  const int bn   = blockIdx.x * 128;

  v8f acc[2][4];
#pragma unroll
  for (int m = 0; m < 2; ++m)
#pragma unroll
    for (int n = 0; n < 4; ++n) acc[m][n] = v8f_zero();

  for (int k0 = 0; k0 < K; k0 += 64) {
    // cooperative tile load: 1024 chunks of 8 bf16 (16B) each for A and B
#pragma unroll
    for (int i = 0; i < 4; ++i) {
      int c   = tid + 256 * i;
      int row = c >> 3;
      int col = (c & 7) << 3;
      copy16_to_lds(&sA[row * 64 + col], &A[(size_t)(bm + row) * K + k0 + col]);
      copy16_to_lds(&sB[row * 64 + col], &Bt[(size_t)(bn + row) * K + k0 + col]);
    }
    wait_lds_copies();
    __syncthreads();

    v16bf af[2][2], bf[4][2];
#pragma unroll
    for (int m = 0; m < 2; ++m)
#pragma unroll
      for (int ks = 0; ks < 2; ++ks)
        af[m][ks] = load_frag_a(sA, wm * 32 + m * 16, ks * 32, 64, lane);
#pragma unroll
    for (int n = 0; n < 4; ++n)
#pragma unroll
      for (int ks = 0; ks < 2; ++ks)
        bf[n][ks] = load_frag_b(sB, wn * 64 + n * 16, ks * 32, 64, lane);
#pragma unroll
    for (int m = 0; m < 2; ++m)
#pragma unroll
      for (int n = 0; n < 4; ++n) {
        acc[m][n] = wmma_bf16(af[m][0], bf[n][0], acc[m][n]);
        acc[m][n] = wmma_bf16(af[m][1], bf[n][1], acc[m][n]);
      }
    __syncthreads();
  }

  const int hf   = lane >> 4;
  const int lcol = lane & 15;
#pragma unroll
  for (int m = 0; m < 2; ++m)
#pragma unroll
    for (int n = 0; n < 4; ++n) {
      size_t base = (size_t)(bm + wm * 32 + m * 16 + hf * 8) * N +
                    (bn + wn * 64 + n * 16 + lcol);
#pragma unroll
      for (int vr = 0; vr < 8; ++vr) {
        if constexpr (F32OUT)
          ((float*)Cout)[base + (size_t)vr * N] = acc[m][n][vr];
        else
          ((__bf16*)Cout)[base + (size_t)vr * N] = f2bf(acc[m][n][vr]);
      }
    }
}

// ---------------------------------------------------------------------------
// 4) RMSNorm (full E) + RoPE (per head, adjacent-pair rotate, dup'd angles)
//    One block per token row; handles q and k slices of qkv.
// ---------------------------------------------------------------------------
__global__ __launch_bounds__(256)
void normrope_kernel(const __bf16* __restrict__ qkv, const float* __restrict__ lnw,
                     __bf16* __restrict__ qn, __bf16* __restrict__ kn) {
  const int r = blockIdx.x;         // token row 0..M_TOT-1
  const int s = r & (LSEQ - 1);     // sequence position
  const int t = threadIdx.x;
  const int e0 = t * 4;
  const size_t base = (size_t)r * E3;

  union { uint2 u; __bf16 e[4]; } qld, kld;
  qld.u = *(const uint2*)(&qkv[base + e0]);
  kld.u = *(const uint2*)(&qkv[base + E_DIM + e0]);

  float qv[4], kv[4];
  float ssq = 0.f, ssk = 0.f;
#pragma unroll
  for (int i = 0; i < 4; ++i) {
    qv[i] = bf2f(qld.e[i]);
    kv[i] = bf2f(kld.e[i]);
    ssq += qv[i] * qv[i];
    ssk += kv[i] * kv[i];
  }
  __shared__ float redq[8], redk[8];
#pragma unroll
  for (int off = 16; off > 0; off >>= 1) {
    ssq += __shfl_xor(ssq, off, 32);
    ssk += __shfl_xor(ssk, off, 32);
  }
  if ((t & 31) == 0) { redq[t >> 5] = ssq; redk[t >> 5] = ssk; }
  __syncthreads();
  ssq = 0.f; ssk = 0.f;
#pragma unroll
  for (int i = 0; i < 8; ++i) { ssq += redq[i]; ssk += redk[i]; }
  const float rq = rsqrtf(ssq * (1.0f / E_DIM) + 1e-6f);
  const float rk = rsqrtf(ssk * (1.0f / E_DIM) + 1e-6f);

  float qw[4], kw[4];
#pragma unroll
  for (int i = 0; i < 4; ++i) {
    float w = lnw[e0 + i];
    qw[i] = qv[i] * rq * w;
    kw[i] = kv[i] * rk * w;
  }

  union { uint2 u; __bf16 e[4]; } qst, kst;
#pragma unroll
  for (int p = 0; p < 2; ++p) {
    int de = (e0 + 2 * p) & (HDIM - 1);  // even element's d within head
    float fe = __powf(10000.f, -(float)(4 * (de & 31) + 1) * (1.0f / HDIM));
    float fo = __powf(10000.f, -(float)(4 * ((de + 1) & 31) + 1) * (1.0f / HDIM));
    float ae = (float)s * fe, ao = (float)s * fo;
    float ce = __cosf(ae), se = __sinf(ae);
    float co = __cosf(ao), so = __sinf(ao);
    float qe = qw[2 * p], qo = qw[2 * p + 1];
    float ke = kw[2 * p], ko = kw[2 * p + 1];
    qst.e[2 * p]     = f2bf(qe * ce - qo * se);
    qst.e[2 * p + 1] = f2bf(qo * co + qe * so);
    kst.e[2 * p]     = f2bf(ke * ce - ko * se);
    kst.e[2 * p + 1] = f2bf(ko * co + ke * so);
  }
  const size_t ob = (size_t)r * E_DIM + e0;
  *(uint2*)(&qn[ob]) = qst.u;
  *(uint2*)(&kn[ob]) = kst.u;
}

// ---------------------------------------------------------------------------
// 5) Flash attention: one block per (head, 128-row q block). 8 waves,
//    each owns 16 q rows. Key blocks of 64, causal-bounded, online softmax.
// ---------------------------------------------------------------------------
__global__ __launch_bounds__(256)
void flash_attn_kernel(const __bf16* __restrict__ Q, const __bf16* __restrict__ Kc,
                       const __bf16* __restrict__ QKV, __bf16* __restrict__ Y) {
  __shared__ __bf16 sQ[128 * 64];   // 16 KB
  __shared__ __bf16 sK[64 * 64];    //  8 KB
  __shared__ __bf16 sVt[64 * 64];   //  8 KB (V transposed: [d][key])
  __shared__ __bf16 sP[128 * 64];   // 16 KB

  const int bh   = blockIdx.x;      // 0..255 = (b*NG)*NH + h
  const int qb   = blockIdx.y;      // 0..3
  const int bg   = bh >> 4;
  const int h    = bh & 15;
  const int hoff = h * HDIM;
  const size_t rowbase = (size_t)bg * LSEQ;
  const int s0   = qb * 128;

  const int tid  = threadIdx.x;
  const int lane = tid & 31;
  const int wave = tid >> 5;
  const int hf   = lane >> 4;
  const int lcol = lane & 15;

  // Load 128x64 Q block (bf16, post norm+rope)
#pragma unroll
  for (int i = 0; i < 2; ++i) {
    int c   = tid * 2 + i;          // 512 chunks of 16 elems
    int row = c >> 2;
    int d0  = (c & 3) << 4;
    const __bf16* src = Q + (rowbase + s0 + row) * E_DIM + hoff + d0;
    copy16_to_lds(&sQ[row * 64 + d0],     src);
    copy16_to_lds(&sQ[row * 64 + d0 + 8], src + 8);
  }
  wait_lds_copies();
  __syncthreads();

  v16bf qf[2];
  qf[0] = load_frag_a(sQ, wave * 16, 0, 64, lane);
  qf[1] = load_frag_a(sQ, wave * 16, 32, 64, lane);

  float m_run[8], l_run[8];
  v8f o_acc[4];
#pragma unroll
  for (int vr = 0; vr < 8; ++vr) { m_run[vr] = -1e30f; l_run[vr] = 0.f; }
#pragma unroll
  for (int nt = 0; nt < 4; ++nt) o_acc[nt] = v8f_zero();

  const int nkb = 2 * qb + 2;       // causal: keys up to (qb+1)*128
  for (int kb = 0; kb < nkb; ++kb) {
    __syncthreads();                // waves done with previous sK/sVt
    {
      int row = tid >> 2;           // 64 rows x 4 chunks of 16
      int d0  = (tid & 3) << 4;
      const __bf16* ksrc = Kc + (rowbase + kb * 64 + row) * E_DIM + hoff + d0;
      copy16_to_lds(&sK[row * 64 + d0],     ksrc);
      copy16_to_lds(&sK[row * 64 + d0 + 8], ksrc + 8);
      // V slice of raw qkv (cols 2E..3E): vector load, transpose into sVt
      const __bf16* vsrc = QKV + (rowbase + kb * 64 + row) * E3 + 2 * E_DIM + hoff + d0;
      union { uint4 u[2]; __bf16 e[16]; } vb;
      vb.u[0] = *(const uint4*)(vsrc);
      vb.u[1] = *(const uint4*)(vsrc + 8);
#pragma unroll
      for (int j = 0; j < 16; ++j) sVt[(d0 + j) * 64 + row] = vb.e[j];
    }
    wait_lds_copies();
    __syncthreads();

    // S = Q * K^T  (per wave: 16 x 64 scores)
    v8f sc[4];
#pragma unroll
    for (int nt = 0; nt < 4; ++nt) {
      v8f a = v8f_zero();
      a = wmma_bf16(qf[0], load_frag_b(sK, nt * 16, 0, 64, lane), a);
      a = wmma_bf16(qf[1], load_frag_b(sK, nt * 16, 32, 64, lane), a);
      sc[nt] = a;
    }

    // online softmax; row m = wave*16 + vr + 8*hf lives in one 16-lane half
#pragma unroll
    for (int vr = 0; vr < 8; ++vr) {
      int qi = s0 + wave * 16 + vr + hf * 8;
      float rowmax = -1e30f;
#pragma unroll
      for (int nt = 0; nt < 4; ++nt) {
        int kj  = kb * 64 + nt * 16 + lcol;
        float v = sc[nt][vr] * 0.125f;            // 1/sqrt(64)
        v = (kj > qi) ? -1e30f : v;
        sc[nt][vr] = v;
        rowmax = fmaxf(rowmax, v);
      }
#pragma unroll
      for (int off = 1; off < 16; off <<= 1)
        rowmax = fmaxf(rowmax, __shfl_xor(rowmax, off, 32));
      float newm = fmaxf(m_run[vr], rowmax);
      float fs   = __expf(m_run[vr] - newm);
      float rsum = 0.f;
#pragma unroll
      for (int nt = 0; nt < 4; ++nt) {
        float p = __expf(sc[nt][vr] - newm);
        sc[nt][vr] = p;
        rsum += p;
      }
#pragma unroll
      for (int off = 1; off < 16; off <<= 1)
        rsum += __shfl_xor(rsum, off, 32);
      m_run[vr] = newm;
      l_run[vr] = l_run[vr] * fs + rsum;
#pragma unroll
      for (int nt = 0; nt < 4; ++nt) o_acc[nt][vr] *= fs;
      // stage P into LDS (own wave strip only) for A-fragment re-layout
#pragma unroll
      for (int nt = 0; nt < 4; ++nt)
        sP[(wave * 16 + vr + hf * 8) * 64 + nt * 16 + lcol] = f2bf(sc[nt][vr]);
    }

    // O += P * V
    v16bf pf[2];
    pf[0] = load_frag_a(sP, wave * 16, 0, 64, lane);
    pf[1] = load_frag_a(sP, wave * 16, 32, 64, lane);
#pragma unroll
    for (int nt = 0; nt < 4; ++nt) {
      o_acc[nt] = wmma_bf16(pf[0], load_frag_b(sVt, nt * 16, 0, 64, lane), o_acc[nt]);
      o_acc[nt] = wmma_bf16(pf[1], load_frag_b(sVt, nt * 16, 32, 64, lane), o_acc[nt]);
    }
  }

  // normalize and store attention output (bf16, head-interleaved [M, E])
#pragma unroll
  for (int vr = 0; vr < 8; ++vr) {
    float inv = 1.0f / l_run[vr];
    size_t row = rowbase + s0 + wave * 16 + vr + hf * 8;
#pragma unroll
    for (int nt = 0; nt < 4; ++nt)
      Y[row * E_DIM + hoff + nt * 16 + lcol] = f2bf(o_acc[nt][vr] * inv);
  }
}

// ---------------------------------------------------------------------------
// Host-side orchestration
// ---------------------------------------------------------------------------
extern "C" void kernel_launch(void* const* d_in, const int* in_sizes, int n_in,
                              void* d_out, int out_size, void* d_ws, size_t ws_size,
                              hipStream_t stream) {
  const float* x      = (const float*)d_in[0];
  const float* W_qkv  = (const float*)d_in[1];
  const float* ln_w   = (const float*)d_in[2];
  const float* W_proj = (const float*)d_in[3];
  float* out = (float*)d_out;

  char* ws = (char*)d_ws;
  size_t off = 0;
  auto carve = [&](size_t bytes) {
    char* p = ws + off;
    off = (off + bytes + 255) & ~(size_t)255;
    return p;
  };
  __bf16* XB     = (__bf16*)carve((size_t)M_TOT * E_DIM * 2);  // x bf16
  __bf16* WQKVT  = (__bf16*)carve((size_t)E3 * E_DIM * 2);     // W_qkv^T bf16
  __bf16* WPROJT = (__bf16*)carve((size_t)E_DIM * E_DIM * 2);  // W_proj^T bf16
  __bf16* QKVB   = (__bf16*)carve((size_t)M_TOT * E3 * 2);     // raw qkv bf16
  __bf16* QN     = (__bf16*)carve((size_t)M_TOT * E_DIM * 2);  // q norm+rope
  __bf16* KN     = (__bf16*)carve((size_t)M_TOT * E_DIM * 2);  // k norm+rope
  __bf16* YATT   = (__bf16*)carve((size_t)M_TOT * E_DIM * 2);  // attn out

  // 1) convert inputs to bf16 (weights transposed for B-fragment loads)
  {
    int n = M_TOT * E_DIM;
    cvt_f32_bf16_kernel<<<n / (256 * 8), 256, 0, stream>>>(x, XB);
  }
  {
    int n = E_DIM * E3;
    transpose_cvt_kernel<<<(n + 255) / 256, 256, 0, stream>>>(W_qkv, WQKVT, E_DIM, E3);
  }
  {
    int n = E_DIM * E_DIM;
    transpose_cvt_kernel<<<(n + 255) / 256, 256, 0, stream>>>(W_proj, WPROJT, E_DIM, E_DIM);
  }

  // 2) QKV projection: [8192,1024] x [1024,3072] -> bf16
  wmma_gemm_kernel<E3, E_DIM, false>
      <<<dim3(E3 / 128, M_TOT / 128), 256, 0, stream>>>(XB, WQKVT, QKVB);

  // 3) RMSNorm + RoPE on q,k
  normrope_kernel<<<M_TOT, 256, 0, stream>>>(QKVB, ln_w, QN, KN);

  // 4) flash attention: 256 heads x 4 q-blocks
  flash_attn_kernel<<<dim3(256, 4), 256, 0, stream>>>(QN, KN, QKVB, YATT);

  // 5) output projection: [8192,1024] x [1024,1024] -> f32 out
  wmma_gemm_kernel<E_DIM, E_DIM, true>
      <<<dim3(E_DIM / 128, M_TOT / 128), 256, 0, stream>>>(YATT, WPROJT, out);
}
